// Attention_19653770346763
// MI455X (gfx1250) — matrix operand order repaired and checked by reading, not compiled, not run on hardware
//
#include <hip/hip_runtime.h>
#include <hip/hip_bf16.h>

typedef __attribute__((ext_vector_type(16))) _Float16 v16h;
typedef __attribute__((ext_vector_type(8)))  _Float16 v8h;
typedef __attribute__((ext_vector_type(4)))  _Float16 v4h;
typedef __attribute__((ext_vector_type(8)))  float    v8f;
typedef __attribute__((ext_vector_type(4)))  int      v4i;

#define B      4
#define NSEQ   2048
#define DIM    1024
#define HEADS  16
#define DHEAD  64
#define INNER  1024
#define MROWS  (B * NSEQ)   // 8192

#if defined(__has_builtin)
#if __has_builtin(__builtin_amdgcn_global_load_async_to_lds_b128)
#define HAVE_ASYNC_LDS 1
#endif
#endif

// ---------------------------------------------------------------------------
// WMMA helper: D = A(16x32 f16) * B(32x16 f16) + C(16x16 f32)
// ---------------------------------------------------------------------------
__device__ inline v8f wmma_f16(v16h a, v16h b, v8f c) {
  return __builtin_amdgcn_wmma_f32_16x16x32_f16(
      /*neg_a=*/false, a, /*neg_b=*/false, b,
      /*c_mod=*/(short)0, c, /*reuse_a=*/false, /*reuse_b=*/false);
}

// A-fragment loader: 16x32 f16 tile, row-major, `stride` elements per row.
//   lanes 0-15  : row = lane,    K = 0..7  and 16..23
//   lanes 16-31 : row = lane-16, K = 8..15 and 24..31
__device__ inline v16h load_fragA(const _Float16* base, int stride) {
  int lane = threadIdx.x & 31;
  int row  = lane & 15;
  int koff = (lane >> 4) << 3;                 // 0 or 8
  const _Float16* p = base + (size_t)row * stride + koff;
  v8h lo = *(const v8h*)(p);
  v8h hi = *(const v8h*)(p + 16);
  v16h r;
#pragma unroll
  for (int i = 0; i < 8; ++i) { r[i] = lo[i]; r[i + 8] = hi[i]; }
  return r;
}

// B-fragment loader from transposed storage (BT: N rows x K contiguous).
//   lanes 0-15  : col = lane,    K = 0..7  and 16..23
//   lanes 16-31 : col = lane-16, K = 8..15 and 24..31
__device__ inline v16h load_fragB(const _Float16* baseT, int stride) {
  int lane = threadIdx.x & 31;
  int col  = lane & 15;
  int k0   = (lane >> 4) << 3;                 // 0 or 8
  const _Float16* p = baseT + (size_t)col * stride + k0;
  v8h lo = *(const v8h*)(p);
  v8h hi = *(const v8h*)(p + 16);
  v16h r;
#pragma unroll
  for (int i = 0; i < 8; ++i) { r[i] = lo[i]; r[i + 8] = hi[i]; }
  return r;
}

// ---------------------------------------------------------------------------
// Kernel 1: LayerNorm (f32 in) -> f16 xn.  One block per row of 1024.
// ---------------------------------------------------------------------------
__global__ __launch_bounds__(256)
void ln_kernel(const float* __restrict__ x, const float* __restrict__ gam,
               const float* __restrict__ bet, _Float16* __restrict__ xnh) {
  __shared__ float red[16];
  const int row = blockIdx.x;
  const int tid = threadIdx.x;
  const float4 v = ((const float4*)(x + (size_t)row * DIM))[tid];
  float s  = v.x + v.y + v.z + v.w;
  float ss = v.x * v.x + v.y * v.y + v.z * v.z + v.w * v.w;
#pragma unroll
  for (int off = 16; off; off >>= 1) {
    s  += __shfl_xor(s,  off, 32);
    ss += __shfl_xor(ss, off, 32);
  }
  if ((tid & 31) == 0) {
    red[(tid >> 5) * 2 + 0] = s;
    red[(tid >> 5) * 2 + 1] = ss;
  }
  __syncthreads();
  float ts = 0.f, tss = 0.f;
#pragma unroll
  for (int w = 0; w < 8; ++w) { ts += red[w * 2]; tss += red[w * 2 + 1]; }
  const float mean = ts * (1.0f / DIM);
  const float var  = tss * (1.0f / DIM) - mean * mean;
  const float rstd = rsqrtf(var + 1e-5f);
  const float4 gv = ((const float4*)gam)[tid];
  const float4 bv = ((const float4*)bet)[tid];
  v4h o;
  o[0] = (_Float16)((v.x - mean) * rstd * gv.x + bv.x);
  o[1] = (_Float16)((v.y - mean) * rstd * gv.y + bv.y);
  o[2] = (_Float16)((v.z - mean) * rstd * gv.z + bv.z);
  o[3] = (_Float16)((v.w - mean) * rstd * gv.w + bv.w);
  ((v4h*)(xnh + (size_t)row * DIM))[tid] = o;
}

// ---------------------------------------------------------------------------
// Kernel 2: weight convert f32 [K][N] -> f16 transposed [N][K], with scale.
// ---------------------------------------------------------------------------
__global__ __launch_bounds__(256)
void wcvt_kernel(const float* __restrict__ W, _Float16* __restrict__ WT,
                 int K, int N, float scale) {
  int idx = blockIdx.x * 256 + threadIdx.x;
  if (idx >= K * N) return;
  int k = idx / N;
  int n = idx - k * N;
  WT[(size_t)n * K + k] = (_Float16)(W[idx] * scale);
}

// ---------------------------------------------------------------------------
// Kernel 3: generic f16 WMMA GEMM.  C[M x N] = A[M x K] * BT[N x K]^T.
// Block = 8 waves -> 128 x 64 tile; wave = 16 x 64 strip (A reused x4).
// B slab (64 x 32 f16 = 4KB) is shared by all 8 waves: staged through LDS
// with GLOBAL_LOAD_ASYNC_TO_LDS_B128 + s_wait_asynccnt (double-buffered)
// when available; otherwise register-double-buffered direct loads.
// Output modes:
//   0: f32 row-major (stride N) into outF
//   1: f16 scatter into q layout  [b][h][n][64]      (outH)
//   2: cols < INNER -> k layout [b][h][n][64] (outH);
//      cols >= INNER -> vT layout [b][h][64][n] (outH2)
// ---------------------------------------------------------------------------
__global__ __launch_bounds__(256)
void gemm_kernel(const _Float16* __restrict__ A, const _Float16* __restrict__ BT,
                 float* __restrict__ outF, _Float16* __restrict__ outH,
                 _Float16* __restrict__ outH2, int M, int N, int K, int mode) {
  const int tid  = threadIdx.x;
  const int lane = tid & 31;
  const int wave = tid >> 5;
  const int m0 = blockIdx.x * 128 + wave * 16;
  const int n0 = blockIdx.y * 64;

  v8f acc[4] = {};
  const _Float16* Ab = A + (size_t)m0 * K;

#if defined(HAVE_ASYNC_LDS)
  // -------- async-to-LDS double-buffered B staging --------
  __shared__ _Float16 bsh[2][64 * 32];
  typedef __attribute__((address_space(1))) v4i gv4i;
  typedef __attribute__((address_space(3))) v4i lv4i;
  const int brow = tid >> 2;          // 0..63  (BT row within slab)
  const int bseg = (tid & 3) * 8;     // 8-half (16B) segment within 32-k row
  auto issueB = [&](int buf, int k) {
    const _Float16* src = BT + (size_t)(n0 + brow) * K + k + bseg;
    _Float16* dst = &bsh[buf][brow * 32 + bseg];
    __builtin_amdgcn_global_load_async_to_lds_b128((gv4i*)src, (lv4i*)dst, 0, 0);
  };
  issueB(0, 0);
  v16h a = load_fragA(Ab, K);
  for (int k = 0; k < K; k += 32) {
    const int cur = (k >> 5) & 1;
    asm volatile("s_wait_asynccnt 0" ::: "memory");
    __syncthreads();                          // slab `cur` ready for all waves
    if (k + 32 < K) issueB(cur ^ 1, k + 32);  // prefetch next slab
    v16h an = (k + 32 < K) ? load_fragA(Ab + k + 32, K) : a;
#pragma unroll
    for (int t = 0; t < 4; ++t) {
      v16h bm = load_fragB(&bsh[cur][t * 16 * 32], 32);
      acc[t] = wmma_f16(a, bm, acc[t]);
    }
    a = an;
    __syncthreads();                          // readers done before overwrite
  }
#else
  // -------- register double-buffered direct loads --------
  v16h a = load_fragA(Ab, K);
  v16h bm[4];
#pragma unroll
  for (int t = 0; t < 4; ++t)
    bm[t] = load_fragB(BT + (size_t)(n0 + t * 16) * K, K);
  for (int k = 0; k < K; k += 32) {
    v16h an = a;
    v16h bn[4];
    if (k + 32 < K) {
      an = load_fragA(Ab + k + 32, K);
#pragma unroll
      for (int t = 0; t < 4; ++t)
        bn[t] = load_fragB(BT + (size_t)(n0 + t * 16) * K + k + 32, K);
    }
#pragma unroll
    for (int t = 0; t < 4; ++t) acc[t] = wmma_f16(a, bm[t], acc[t]);
    a = an;
#pragma unroll
    for (int t = 0; t < 4; ++t) bm[t] = bn[t];
  }
#endif

  const int g  = lane >> 4;
  const int nn = lane & 15;
#pragma unroll
  for (int t = 0; t < 4; ++t) {
    const int col = n0 + t * 16 + nn;
#pragma unroll
    for (int r = 0; r < 8; ++r) {
      const int m = m0 + r + 8 * g;
      const float val = acc[t][r];
      if (mode == 0) {
        outF[(size_t)m * N + col] = val;
      } else {
        const int bb = m >> 11;          // / NSEQ
        const int nq = m & (NSEQ - 1);
        if (mode == 1 || col < INNER) {
          const int c = col;
          const int hh = c >> 6, d = c & 63;
          outH[((((size_t)bb * HEADS + hh) * NSEQ) + nq) * DHEAD + d] =
              (_Float16)val;
        } else {
          const int c = col - INNER;
          const int hh = c >> 6, d = c & 63;
          outH2[(((size_t)bb * HEADS + hh) * DHEAD + d) * NSEQ + nq] =
              (_Float16)val;
        }
      }
    }
  }
}

// ---------------------------------------------------------------------------
// Kernel 4: flash attention.  One wave (32 threads) per 16-query tile.
// Per 32-key step: S = Q K^T (4 WMMAs) with K fragments prefetched one step
// ahead; V fragments + bias issued BEFORE the softmax VALU work so the
// exp/shuffle math hides their latency; P bounced through LDS with an
// explicit s_wait_dscnt; O += P V (4 WMMAs).
// ---------------------------------------------------------------------------
__global__ __launch_bounds__(32)
void attn_kernel(const _Float16* __restrict__ qh, const _Float16* __restrict__ kh,
                 const _Float16* __restrict__ vT, const float* __restrict__ bias,
                 _Float16* __restrict__ aout) {
  __shared__ _Float16 pbuf[16 * 32];
  const int lane = threadIdx.x & 31;
  const int tile = blockIdx.x;
  const int q0 = (tile & 127) << 4;
  const int h  = (tile >> 7) & 15;
  const int b  = tile >> 11;

  const _Float16* qb = qh + (((size_t)(b * HEADS + h) * NSEQ) + q0) * DHEAD;
  const _Float16* kb = kh + ((size_t)(b * HEADS + h) * NSEQ) * DHEAD;
  const _Float16* vb = vT + ((size_t)(b * HEADS + h) * DHEAD) * NSEQ;
  const float*    bb = bias + ((size_t)h * NSEQ + q0) * NSEQ;

  const v16h aq0 = load_fragA(qb + 0,  DHEAD);   // q, d = 0..31  (pre-scaled)
  const v16h aq1 = load_fragA(qb + 32, DHEAD);   // q, d = 32..63

  v8f o[4] = {};
  float rmax[8], rsum[8];
#pragma unroll
  for (int r = 0; r < 8; ++r) { rmax[r] = -1e30f; rsum[r] = 0.0f; }

  const int g  = lane >> 4;
  const int nn = lane & 15;

  // preload K fragments for kt = 0
  v16h bk[4];
  bk[0] = load_fragB(kb + 0 * DHEAD + 0,  DHEAD);
  bk[1] = load_fragB(kb + 0 * DHEAD + 32, DHEAD);
  bk[2] = load_fragB(kb + 16 * DHEAD + 0,  DHEAD);
  bk[3] = load_fragB(kb + 16 * DHEAD + 32, DHEAD);

  for (int kt = 0; kt < NSEQ; kt += 32) {
    // ---- S = Q * K^T for 32 keys (two 16x16 C tiles) ----
    v8f s0 = {}, s1 = {};
    s0 = wmma_f16(aq0, bk[0], s0);
    s0 = wmma_f16(aq1, bk[1], s0);
    s1 = wmma_f16(aq0, bk[2], s1);
    s1 = wmma_f16(aq1, bk[3], s1);

    // ---- issue V fragments (current step) + next K fragments + bias ----
    v16h bv[4];
#pragma unroll
    for (int t = 0; t < 4; ++t)
      bv[t] = load_fragB(vb + (size_t)(t * 16) * NSEQ + kt, NSEQ);
    if (kt + 32 < NSEQ) {
      const _Float16* nkb = kb + (size_t)(kt + 32) * DHEAD;
      bk[0] = load_fragB(nkb + 0,  DHEAD);
      bk[1] = load_fragB(nkb + 32, DHEAD);
      bk[2] = load_fragB(nkb + 16 * DHEAD + 0,  DHEAD);
      bk[3] = load_fragB(nkb + 16 * DHEAD + 32, DHEAD);
    }
    float bx0[8], bx1[8];
#pragma unroll
    for (int r = 0; r < 8; ++r) {
      const float* bp = bb + (size_t)(r + 8 * g) * NSEQ + kt + nn;
      bx0[r] = bp[0];
      bx1[r] = bp[16];
    }

    // ---- online softmax per row (overlaps outstanding loads) ----
#pragma unroll
    for (int r = 0; r < 8; ++r) {
      const int m = r + 8 * g;
      float x0 = s0[r] + bx0[r];
      float x1 = s1[r] + bx1[r];
      float tm = fmaxf(x0, x1);
#pragma unroll
      for (int off = 8; off; off >>= 1) tm = fmaxf(tm, __shfl_xor(tm, off, 16));
      const float nmax  = fmaxf(rmax[r], tm);
      const float alpha = __expf(rmax[r] - nmax);
      rmax[r] = nmax;
      const float p0 = __expf(x0 - nmax);
      const float p1 = __expf(x1 - nmax);
      float tsum = p0 + p1;
#pragma unroll
      for (int off = 8; off; off >>= 1) tsum += __shfl_xor(tsum, off, 16);
      rsum[r] = rsum[r] * alpha + tsum;
#pragma unroll
      for (int t = 0; t < 4; ++t) o[t][r] *= alpha;
      pbuf[m * 32 + nn]      = (_Float16)p0;
      pbuf[m * 32 + 16 + nn] = (_Float16)p1;
    }
    // C-layout -> A-layout bounce through LDS; wait for DS stores to land.
    asm volatile("s_wait_dscnt 0" ::: "memory");
    const v16h ap = load_fragA(pbuf, 32);
    // ---- O += P * V (four 16-wide d chunks) ----
#pragma unroll
    for (int t = 0; t < 4; ++t) o[t] = wmma_f16(ap, bv[t], o[t]);
  }

  // ---- normalize and store f16 into [b][n][h*64] for the output GEMM ----
#pragma unroll
  for (int r = 0; r < 8; ++r) {
    const int m = r + 8 * g;
    const float inv = 1.0f / rsum[r];
    _Float16* op = aout + ((size_t)(b * NSEQ + q0 + m)) * INNER + h * DHEAD + nn;
#pragma unroll
    for (int t = 0; t < 4; ++t) op[t * 16] = (_Float16)(o[t][r] * inv);
  }
}

// ---------------------------------------------------------------------------
// Launch
// ---------------------------------------------------------------------------
extern "C" void kernel_launch(void* const* d_in, const int* in_sizes, int n_in,
                              void* d_out, int out_size, void* d_ws, size_t ws_size,
                              hipStream_t stream) {
  const float* x    = (const float*)d_in[0];
  const float* bias = (const float*)d_in[1];
  const float* ln_g = (const float*)d_in[2];
  const float* ln_b = (const float*)d_in[3];
  const float* Wq   = (const float*)d_in[4];
  const float* Wkv  = (const float*)d_in[5];
  const float* Wout = (const float*)d_in[6];
  float* out = (float*)d_out;

  char* ws = (char*)d_ws;
  size_t off = 0;
  _Float16* xnh  = (_Float16*)(ws + off); off += (size_t)MROWS * DIM * 2;          // 16 MB
  _Float16* wqT  = (_Float16*)(ws + off); off += (size_t)INNER * DIM * 2;          //  2 MB
  _Float16* wkvT = (_Float16*)(ws + off); off += (size_t)2 * INNER * DIM * 2;      //  4 MB
  _Float16* woT  = (_Float16*)(ws + off); off += (size_t)DIM * INNER * 2;          //  2 MB
  _Float16* qhp  = (_Float16*)(ws + off); off += (size_t)MROWS * INNER * 2;        // 16 MB
  _Float16* khp  = (_Float16*)(ws + off); off += (size_t)MROWS * INNER * 2;        // 16 MB
  _Float16* vTp  = (_Float16*)(ws + off); off += (size_t)MROWS * INNER * 2;        // 16 MB
  _Float16* aoh  = (_Float16*)(ws + off); off += (size_t)MROWS * INNER * 2;        // 16 MB

  // 1) LayerNorm -> f16
  ln_kernel<<<MROWS, 256, 0, stream>>>(x, ln_g, ln_b, xnh);

  // 2) weight conversion (q scale = dhead^-0.5 folded into WqT)
  wcvt_kernel<<<(DIM * INNER + 255) / 256, 256, 0, stream>>>(Wq, wqT, DIM, INNER, 0.125f);
  wcvt_kernel<<<(DIM * 2 * INNER + 255) / 256, 256, 0, stream>>>(Wkv, wkvT, DIM, 2 * INNER, 1.0f);
  wcvt_kernel<<<(INNER * DIM + 255) / 256, 256, 0, stream>>>(Wout, woT, INNER, DIM, 1.0f);

  // 3) q = xn @ Wq  (scatter into [b,h,n,64])
  {
    dim3 grid(MROWS / 128, INNER / 64);
    gemm_kernel<<<grid, 256, 0, stream>>>(xnh, wqT, nullptr, qhp, nullptr,
                                          MROWS, INNER, DIM, 1);
  }
  // 4) kv = xn @ Wkv  (k -> [b,h,n,64]; v -> [b,h,64,n] transposed)
  {
    dim3 grid(MROWS / 128, (2 * INNER) / 64);
    gemm_kernel<<<grid, 256, 0, stream>>>(xnh, wkvT, nullptr, khp, vTp,
                                          MROWS, 2 * INNER, DIM, 2);
  }
  // 5) flash attention: one wave per 16-query tile
  attn_kernel<<<B * HEADS * (NSEQ / 16), 32, 0, stream>>>(qhp, khp, vTp, bias, aoh);

  // 6) out = attn_out @ Wout  -> f32
  {
    dim3 grid(MROWS / 128, DIM / 64);
    gemm_kernel<<<grid, 256, 0, stream>>>(aoh, woT, out, nullptr, nullptr,
                                          MROWS, DIM, INNER, 0);
  }
}
